// SocialPooling_223338299637
// MI455X (gfx1250) — compile-verified, matrix operand order checked by reading, not run on hardware
//
#include <hip/hip_runtime.h>

// ---------------- problem constants (from reference) ----------------
#define H_DIM   64
#define GRID_N  8
#define G2      (GRID_N * GRID_N)      // 64 cells
#define FEAT    (G2 * H_DIM)           // 4096 = K of the GEMM

// ---------------- CDNA5 WMMA types ----------------
typedef __attribute__((ext_vector_type(16))) __bf16        v16bf;
typedef __attribute__((ext_vector_type(8)))  float         v8f;
typedef __attribute__((ext_vector_type(4)))  unsigned int  u32x4;

union Frag {              // 32 bytes: one 16x32 bf16 A/B fragment per lane
    v16bf v;
    u32x4 q[2];
};

struct FragSet {          // A fragment + 4 B fragments (one 16x64 output strip)
    Frag a, b0, b1, b2, b3;
};

__device__ __forceinline__ unsigned short f32_to_bf16(float f) {
    unsigned u = __builtin_bit_cast(unsigned, f);
    u += 0x7FFFu + ((u >> 16) & 1u);          // round-to-nearest-even
    return (unsigned short)(u >> 16);
}

// ---------------------------------------------------------------
// Kernel 1: W (FEAT x N, f32 row-major) -> Wt (N x FEAT, bf16)
// LDS-tiled so both global read and write are coalesced.
// ---------------------------------------------------------------
__global__ void w_transpose_bf16(const float* __restrict__ W,
                                 unsigned short* __restrict__ Wt, int N) {
    __shared__ float tile[32][33];
    int n0 = blockIdx.x * 32;
    int k0 = blockIdx.y * 32;
    int tx = threadIdx.x;          // 0..31
    int ty = threadIdx.y;          // 0..7
    #pragma unroll
    for (int r = ty; r < 32; r += 8)
        tile[r][tx] = W[(size_t)(k0 + r) * N + (n0 + tx)];
    __syncthreads();
    #pragma unroll
    for (int r = ty; r < 32; r += 8)
        Wt[(size_t)(n0 + r) * FEAT + (k0 + tx)] = f32_to_bf16(tile[tx][r]);
}

// ---------------------------------------------------------------
// Kernel 2: social pooling. One 64-thread block per pooling center.
// Cell index is uniform across the 64 h-lanes for a given neighbor j,
// and thread t only ever touches LDS words congruent to t mod 64,
// so the accumulation is race-free without atomics.
// Emits the pooled row directly as bf16 (WMMA A-matrix operand).
// ---------------------------------------------------------------
__global__ void social_pool_bf16(const float* __restrict__ hidden,   // (S*P, 64)
                                 const float* __restrict__ pos,      // (S*P, 2)
                                 unsigned short* __restrict__ Apool, // (S*P, FEAT)
                                 int P) {
    __shared__ float buf[FEAT];    // 16 KB accumulation grid (64 cells x 64 dims)
    __shared__ float px[64], py[64];

    int s = blockIdx.x / P;
    int i = blockIdx.x % P;
    int t = threadIdx.x;           // 0..63  (h-dim lane)
    int base = s * P;

    px[t] = pos[(size_t)(base + t) * 2 + 0];
    py[t] = pos[(size_t)(base + t) * 2 + 1];
    #pragma unroll
    for (int c = 0; c < G2; ++c) buf[c * H_DIM + t] = 0.0f;
    __syncthreads();

    float xi = px[i], yi = py[i];
    for (int j = 0; j < P; ++j) {
        float dx = px[j] - xi;
        float dy = py[j] - yi;
        // strict inequalities: boundary-equal points are OOB in the reference
        if (dx > -1.0f && dx < 1.0f && dy > -1.0f && dy < 1.0f && j != i) {
            int cx = (int)floorf((dx + 1.0f) * 4.0f);   // (dx - (-NS/2)) / NS * GRID
            int cy = (int)floorf((1.0f - dy) * 4.0f);   // (NS/2 - dy) / NS * GRID
            int cell = cy * GRID_N + cx;
            buf[cell * H_DIM + t] += hidden[(size_t)(base + j) * H_DIM + t];
        }
    }
    // each thread owns lane t of every cell; no extra barrier needed
    unsigned short* row = Apool + (size_t)(base + i) * FEAT;
    #pragma unroll
    for (int e = 0; e < G2; ++e)
        row[e * H_DIM + t] = f32_to_bf16(buf[e * H_DIM + t]);
}

// ---------------------------------------------------------------
// Kernel 3: out = relu(A @ Wt^T + b) with v_wmma_f32_16x16x32_bf16.
// A: (M, FEAT) bf16 row-major.  Bt: (N, FEAT) bf16 row-major (= W^T),
// so the B-fragment's per-lane column gather is two contiguous B128 loads.
// One wave -> 16x64 output strip; explicit register double-buffering so
// next K-step's 10 loads are in flight while the 4 WMMAs execute
// (partial s_wait_loadcnt instead of draining to 0 before every WMMA).
//
// 16-bit fragment layout (ISA 7.12.2): lane L holds row/col (L&15);
// lanes 0-15 take K-chunks {0..7, 16..23}, lanes 16-31 {8..15, 24..31}.
// ---------------------------------------------------------------
__device__ __forceinline__ void load_set(FragSet& s,
                                         const unsigned short* aRow,
                                         const unsigned short* bRow0,
                                         const unsigned short* bRow1,
                                         const unsigned short* bRow2,
                                         const unsigned short* bRow3,
                                         int k0) {
    s.a.q[0]  = *(const u32x4*)(aRow  + k0);
    s.a.q[1]  = *(const u32x4*)(aRow  + k0 + 16);
    s.b0.q[0] = *(const u32x4*)(bRow0 + k0);
    s.b0.q[1] = *(const u32x4*)(bRow0 + k0 + 16);
    s.b1.q[0] = *(const u32x4*)(bRow1 + k0);
    s.b1.q[1] = *(const u32x4*)(bRow1 + k0 + 16);
    s.b2.q[0] = *(const u32x4*)(bRow2 + k0);
    s.b2.q[1] = *(const u32x4*)(bRow2 + k0 + 16);
    s.b3.q[0] = *(const u32x4*)(bRow3 + k0);
    s.b3.q[1] = *(const u32x4*)(bRow3 + k0 + 16);
}

__device__ __forceinline__ void mma_set(const FragSet& s,
                                        v8f& acc0, v8f& acc1, v8f& acc2, v8f& acc3) {
    acc0 = __builtin_amdgcn_wmma_f32_16x16x32_bf16(false, s.a.v, false, s.b0.v,
                                                   (short)0, acc0, false, false);
    acc1 = __builtin_amdgcn_wmma_f32_16x16x32_bf16(false, s.a.v, false, s.b1.v,
                                                   (short)0, acc1, false, false);
    acc2 = __builtin_amdgcn_wmma_f32_16x16x32_bf16(false, s.a.v, false, s.b2.v,
                                                   (short)0, acc2, false, false);
    acc3 = __builtin_amdgcn_wmma_f32_16x16x32_bf16(false, s.a.v, false, s.b3.v,
                                                   (short)0, acc3, false, false);
}

__global__ void gemm_bf16_wmma(const unsigned short* __restrict__ A,
                               const unsigned short* __restrict__ Bt,
                               const float* __restrict__ bias,
                               float* __restrict__ out,
                               int M, int N) {
    const int K = FEAT;
    int wave   = (blockIdx.x * blockDim.x + threadIdx.x) >> 5;
    int lane   = threadIdx.x & 31;
    int nGroups = N >> 6;                       // 64 output cols per wave
    int mTile  = wave / nGroups;                // 4 waves/block share one mTile
    int nGroup = wave % nGroups;
    if (mTile * 16 >= M) return;

    int half   = lane >> 4;                     // K-half select
    int rowSel = lane & 15;                     // A row / B column within tile
    int nBase  = nGroup * 64;

    const unsigned short* aRow  = A  + (size_t)(mTile * 16 + rowSel) * K + half * 8;
    const unsigned short* bRow0 = Bt + (size_t)(nBase +  0 + rowSel) * K + half * 8;
    const unsigned short* bRow1 = Bt + (size_t)(nBase + 16 + rowSel) * K + half * 8;
    const unsigned short* bRow2 = Bt + (size_t)(nBase + 32 + rowSel) * K + half * 8;
    const unsigned short* bRow3 = Bt + (size_t)(nBase + 48 + rowSel) * K + half * 8;

    float bv0 = bias[nBase +  0 + rowSel];
    float bv1 = bias[nBase + 16 + rowSel];
    float bv2 = bias[nBase + 32 + rowSel];
    float bv3 = bias[nBase + 48 + rowSel];

    v8f acc0 = {}, acc1 = {}, acc2 = {}, acc3 = {};

    // ---- software pipeline: two fragment sets in flight ----
    FragSet s0, s1;
    load_set(s0, aRow, bRow0, bRow1, bRow2, bRow3, 0);

    int k0 = 0;
    for (; k0 < K - 64; k0 += 64) {
        load_set(s1, aRow, bRow0, bRow1, bRow2, bRow3, k0 + 32);
        mma_set(s0, acc0, acc1, acc2, acc3);        // overlaps s1's loads
        load_set(s0, aRow, bRow0, bRow1, bRow2, bRow3, k0 + 64);
        mma_set(s1, acc0, acc1, acc2, acc3);        // overlaps s0's loads
    }
    // drain: k0 == K - 64
    load_set(s1, aRow, bRow0, bRow1, bRow2, bRow3, k0 + 32);
    mma_set(s0, acc0, acc1, acc2, acc3);
    mma_set(s1, acc0, acc1, acc2, acc3);

    // C/D layout: VGPR r -> row (r + 8*half), col rowSel
    #pragma unroll
    for (int r = 0; r < 8; ++r) {
        int m = mTile * 16 + half * 8 + r;
        size_t o = (size_t)m * N + nBase + rowSel;
        out[o +  0] = fmaxf(acc0[r] + bv0, 0.0f);
        out[o + 16] = fmaxf(acc1[r] + bv1, 0.0f);
        out[o + 32] = fmaxf(acc2[r] + bv2, 0.0f);
        out[o + 48] = fmaxf(acc3[r] + bv3, 0.0f);
    }
}

// ---------------------------------------------------------------
extern "C" void kernel_launch(void* const* d_in, const int* in_sizes, int n_in,
                              void* d_out, int out_size, void* d_ws, size_t ws_size,
                              hipStream_t stream) {
    const float* h_states = (const float*)d_in[0];   // (1, B, 64) f32
    const float* end_pos  = (const float*)d_in[2];   // (B, 2)     f32
    const float* W        = (const float*)d_in[4];   // (FEAT, N)  f32
    const float* b        = (const float*)d_in[5];   // (N,)       f32

    int B = in_sizes[0] / H_DIM;      // 4096
    int S = in_sizes[1] / 2;          // 64
    int P = B / S;                    // 64
    int N = in_sizes[5];              // 1024

    // workspace layout: [Wt bf16: N*FEAT][Apool bf16: B*FEAT]  (= 40 MB)
    unsigned short* Wt    = (unsigned short*)d_ws;
    unsigned short* Apool = Wt + (size_t)N * FEAT;

    dim3 tb(32, 8);
    dim3 tg(N / 32, FEAT / 32);
    w_transpose_bf16<<<tg, tb, 0, stream>>>(W, Wt, N);

    social_pool_bf16<<<B, H_DIM, 0, stream>>>(h_states, end_pos, Apool, P);

    int waves  = (B / 16) * (N / 64);          // 4096 wave-jobs
    int blocks = waves / 4;                    // 128 threads = 4 wave32 per block
    gemm_bf16_wmma<<<blocks, 128, 0, stream>>>(Apool, Wt, b, (float*)d_out, B, N);
}
